// EncoderBlock_6081673691141
// MI455X (gfx1250) — matrix operand order repaired and checked by reading, not compiled, not run on hardware
//
#include <hip/hip_runtime.h>

typedef _Float16 half_t;
typedef __attribute__((ext_vector_type(16))) _Float16 v16h_t;
typedef __attribute__((ext_vector_type(8)))  _Float16 v8h_t;
typedef __attribute__((ext_vector_type(4)))  _Float16 v4h_t;
typedef __attribute__((ext_vector_type(8)))  float    v8f_t;
typedef __attribute__((ext_vector_type(4)))  float    v4f_t;
typedef __attribute__((ext_vector_type(4)))  unsigned int v4u_t;
typedef __attribute__((ext_vector_type(8)))  int      v8i_t;
typedef __attribute__((ext_vector_type(4)))  int      v4i_t;

#define N_DM 512
#define N_H  8
#define N_DK 64
#define N_DV 64
#define N_DF 2048
#define N_B  4
#define N_S  1024
#define N_T  (N_B * N_S)   // 4096 tokens
#define LN_EPS 1e-5f

// --------------------------- TDM availability ------------------------------
#if defined(__has_builtin)
#if __has_builtin(__builtin_amdgcn_tensor_load_to_lds)
#define HAVE_TDM 1
#endif
#endif
#ifndef HAVE_TDM
#define HAVE_TDM 0
#endif
#if __has_include(<hip/amd_detail/amd_gfx1250_TDM.h>)
#define TDM_ARITY6 1
#else
#define TDM_ARITY6 0
#endif

#if HAVE_TDM
#if TDM_ARITY6
#warning "CDNA5 marker: using 6-arg __builtin_amdgcn_tensor_load_to_lds (amdgpu-toolchain)"
#else
#warning "CDNA5 marker: using 5-arg __builtin_amdgcn_tensor_load_to_lds (ROCm)"
#endif
#else
#warning "CDNA5 marker: TDM builtin unavailable; cooperative LDS staging fallback"
#endif

// LDS anti-bank-conflict padding: +16B per 1024B (4-bank shift per 1KB).
// Rows never straddle a 1KB boundary (row sizes divide or equal 1024B), so
// pad(row_base + c) == padded_row_base + c; the remap is hoisted out of the
// K loop and inner-loop DS addresses stay immediate-offset.
__device__ __forceinline__ int lds_pad(int off) { return off + ((off >> 10) << 4); }
#define PADDED_BYTES(b) ((b) + (((b) >> 10) << 4))

// ---------------------------------------------------------------------------
// Stage one contiguous A tile (16 rows x ROWELEMS, lda == ROWELEMS) into LDS
// with 16B-per-1KB padding.
// TDM path: one 2D tensor DMA from wave 0 (D# per ISA 8.3-8.5, pad_enable=1,
// pad_interval=7 (1024B), pad_amount=3 (16B)); s_wait_tensorcnt + barrier.
// Fallback: cooperative 16B-chunk copy applying the same padding function.
// ---------------------------------------------------------------------------
template <int EB /*bytes/elem*/, int ROWELEMS>
__device__ __forceinline__ void stage_a_tile(const void* gsrc, void* lds) {
#if HAVE_TDM
  if ((threadIdx.x >> 5) == 0) {      // wave-uniform: only wave 0 issues the DMA
    const unsigned int lds_off = (unsigned int)(unsigned long long)(uintptr_t)lds;
    const unsigned long long ga = (unsigned long long)(uintptr_t)gsrc;
    v4u_t g0;
    g0[0] = 1u;                                         // count = 1 valid D#
    g0[1] = lds_off;                                    // lds_addr [63:32]
    g0[2] = (unsigned int)ga;                           // global_addr [95:64]
    g0[3] = ((unsigned int)(ga >> 32) & 0x01FFFFFFu)    // global_addr [120:96]
            | (2u << 30);                               // type = 2 ("image")
    const int dsz = (EB == 1) ? 0 : (EB == 2) ? 1 : 2;  // data_size code
    v8i_t g1;
    g1[0] = (dsz << 16)                                 // mask=0 | data_size
            | (1 << 20)                                 // pad_enable
            | (7 << 22)                                 // pad_interval = 1024B
            | (3 << 25);                                // pad_amount  = 16B
    g1[1] = (ROWELEMS & 0xFFFF) << 16;                  // tensor_dim0 lo16 @bit48
    g1[2] = (ROWELEMS >> 16) | (16 << 16);              // dim0 hi16 | tensor_dim1=16
    g1[3] = (ROWELEMS & 0xFFFF) << 16;                  // tile_dim0 @bit112
    g1[4] = 16;                                         // tile_dim1=16, tile_dim2=0
    g1[5] = ROWELEMS;                                   // tensor_dim0_stride lo32
    g1[6] = 0;
    g1[7] = 0;
    v4i_t g2 = {0, 0, 0, 0};
    v4i_t g3 = {0, 0, 0, 0};
#if TDM_ARITY6
    v8i_t g4 = {0, 0, 0, 0, 0, 0, 0, 0};
    __builtin_amdgcn_tensor_load_to_lds(g0, g1, g2, g3, g4, 0);
#else
    __builtin_amdgcn_tensor_load_to_lds(g0, g1, g2, g3, 0);
#endif
    __builtin_amdgcn_s_wait_tensorcnt(0);
  }
  __syncthreads();
#else
  constexpr int CHUNKS = 16 * ROWELEMS * EB / 16;
  const uint4* s = (const uint4*)gsrc;
  char* d = (char*)lds;
  for (int i = threadIdx.x; i < CHUNKS; i += blockDim.x)
    *(uint4*)(d + lds_pad(i * 16)) = s[i];
  __syncthreads();
#endif
}

// ---------------------------------------------------------------------------
// WMMA fragment builders (wave32, ISA 7.12.2 layouts)
// A 16x32 f16: lane m = lane&15, g = lane>>4; halves {k0+8g..} / {k0+16+8g..}
// B 32x16 f16 (B^T rows): lane n = lane&15; halves {k0+8g..} / {k0+16+8g..}
// C/D 16x16 f32: VGPR r -> M = r + 8*g, N = lane&15.
// ---------------------------------------------------------------------------
__device__ __forceinline__ v16h_t frag_a_f16(const half_t* arow, int k0, int g) {
  v8h_t lo = *(const v8h_t*)(arow + k0 + 8 * g);
  v8h_t hi = *(const v8h_t*)(arow + k0 + 16 + 8 * g);
  return __builtin_shufflevector(lo, hi, 0, 1, 2, 3, 4, 5, 6, 7,
                                         8, 9, 10, 11, 12, 13, 14, 15);
}

__device__ __forceinline__ v16h_t frag_a_f32(const float* arow, int k0, int g) {
  v8f_t lo = *(const v8f_t*)(arow + k0 + 8 * g);
  v8f_t hi = *(const v8f_t*)(arow + k0 + 16 + 8 * g);
  v16h_t r;
#pragma unroll
  for (int i = 0; i < 8; ++i) {
    r[i]     = (half_t)lo[i];
    r[i + 8] = (half_t)hi[i];
  }
  return r;
}

__device__ __forceinline__ v16h_t frag_b(const half_t* brow, int k0, int gb) {
  v8h_t lo = *(const v8h_t*)(brow + k0 + 8 * gb);
  v8h_t hi = *(const v8h_t*)(brow + k0 + 16 + 8 * gb);
  return __builtin_shufflevector(lo, hi, 0, 1, 2, 3, 4, 5, 6, 7,
                                         8, 9, 10, 11, 12, 13, 14, 15);
}

// Tile from padded LDS A (f16); padded row base hoisted out of the K loop.
template <int KK>
__device__ __forceinline__ v8f_t wmma_rowtile_lds(const char* shA, const half_t* Bt,
                                                  int n0, int lane) {
  const int g  = lane >> 4;
  const int mr = lane & 15;
  const int row_bytes = mr * KK * 2;
  const half_t* arow = (const half_t*)(shA + lds_pad(row_bytes));  // hoisted remap
  v8f_t acc = {};
  const half_t* brow = Bt + (size_t)(n0 + mr) * KK;
#pragma unroll 4
  for (int k0 = 0; k0 < KK; k0 += 32) {
    if (k0 + 64 < KK) __builtin_prefetch((const void*)(brow + k0 + 64), 0, 3);
    v16h_t a = frag_a_f16(arow, k0, g);
    v16h_t b = frag_b(brow, k0, g);
    acc = __builtin_amdgcn_wmma_f32_16x16x32_f16(false, a, false, b,
                                                 (short)0, acc, false, false);
  }
  return acc;
}

// Tile with A read directly from global (f16 or f32).  A points at tile row 0.
template <bool A_IS_F32, int KK>
__device__ __forceinline__ v8f_t wmma_rowtile_glb(const void* A, const half_t* Bt,
                                                  int n0, int lane) {
  const int g  = lane >> 4;
  const int mr = lane & 15;
  v8f_t acc = {};
  const half_t* arow16 = (const half_t*)A + (size_t)mr * KK;
  const float*  arow32 = (const float*)A + (size_t)mr * KK;
  const half_t* brow   = Bt + (size_t)(n0 + mr) * KK;
#pragma unroll 4
  for (int k0 = 0; k0 < KK; k0 += 32) {
    if (k0 + 64 < KK) __builtin_prefetch((const void*)(brow + k0 + 64), 0, 3);
    v16h_t a;
    if constexpr (A_IS_F32) a = frag_a_f32(arow32, k0, g);
    else                    a = frag_a_f16(arow16, k0, g);
    v16h_t b = frag_b(brow, k0, g);
    acc = __builtin_amdgcn_wmma_f32_16x16x32_f16(false, a, false, b,
                                                 (short)0, acc, false, false);
  }
  return acc;
}

// ---------------------------------------------------------------------------
// Block reductions
// ---------------------------------------------------------------------------
__device__ __forceinline__ float block_sum(float v, float* sh) {
  const int tid = threadIdx.x;
  sh[tid] = v;
  __syncthreads();
  for (int s = blockDim.x >> 1; s > 0; s >>= 1) {
    if (tid < s) sh[tid] += sh[tid + s];
    __syncthreads();
  }
  float r = sh[0];
  __syncthreads();
  return r;
}

__device__ __forceinline__ float block_max(float v, float* sh) {
  const int tid = threadIdx.x;
  sh[tid] = v;
  __syncthreads();
  for (int s = blockDim.x >> 1; s > 0; s >>= 1) {
    if (tid < s) sh[tid] = fmaxf(sh[tid], sh[tid + s]);
    __syncthreads();
  }
  float r = sh[0];
  __syncthreads();
  return r;
}

// ---------------------------------------------------------------------------
// Weight prep: f32 -> f16 B^T layout (row n = column n of B, length K)
// ---------------------------------------------------------------------------
__global__ void k_transpose_w(const float* __restrict__ W, half_t* __restrict__ out,
                              int K, int N, int qkv) {
  const int total = K * N;
  for (int i = blockIdx.x * blockDim.x + threadIdx.x; i < total;
       i += gridDim.x * blockDim.x) {
    const int n = i / K;
    const int k = i - n * K;
    float v = qkv ? W[((size_t)(n >> 6) * K + k) * 64 + (n & 63)]
                  : W[(size_t)k * N + n];
    out[i] = (half_t)v;
  }
}

// ---------------------------------------------------------------------------
// LayerNorm over DM=512, f32 in -> f16 out.  1 token / block, 128 threads.
// ---------------------------------------------------------------------------
__global__ void k_layernorm16(const float* __restrict__ x, const float* __restrict__ g,
                              const float* __restrict__ b, half_t* __restrict__ out) {
  __shared__ float sh[128];
  const int t   = blockIdx.x;
  const int tid = threadIdx.x;
  const float* row = x + (size_t)t * N_DM;
  v4f_t v = *(const v4f_t*)(row + tid * 4);
  float mu = block_sum(v[0] + v[1] + v[2] + v[3], sh) * (1.0f / N_DM);
  float d0 = v[0] - mu, d1 = v[1] - mu, d2 = v[2] - mu, d3 = v[3] - mu;
  float var = block_sum(d0 * d0 + d1 * d1 + d2 * d2 + d3 * d3, sh) * (1.0f / N_DM);
  float rstd = rsqrtf(var + LN_EPS);
  v4f_t gg = *(const v4f_t*)(g + tid * 4);
  v4f_t bb = *(const v4f_t*)(b + tid * 4);
  v4h_t o;
  o[0] = (half_t)(d0 * rstd * gg[0] + bb[0]);
  o[1] = (half_t)(d1 * rstd * gg[1] + bb[1]);
  o[2] = (half_t)(d2 * rstd * gg[2] + bb[2]);
  o[3] = (half_t)(d3 * rstd * gg[3] + bb[3]);
  *(v4h_t*)(out + (size_t)t * N_DM + tid * 4) = o;
}

// ---------------------------------------------------------------------------
// QKV projection: [T,512] @ Bt[512,512] -> q,k [B,H,S,DK]; v -> [B,H,DV,S].
// grid (T/16, 4, 3), block 256 (8 waves), A tile TDM-staged in LDS.
// ---------------------------------------------------------------------------
__global__ __launch_bounds__(256) void k_qkv(
    const half_t* __restrict__ h16, const half_t* __restrict__ wqt,
    const half_t* __restrict__ wkt, const half_t* __restrict__ wvt,
    const float* __restrict__ qb, const float* __restrict__ kb,
    const float* __restrict__ vb, half_t* __restrict__ q,
    half_t* __restrict__ k, half_t* __restrict__ vt) {
  __shared__ alignas(32) char shA[PADDED_BYTES(16 * N_DM * 2)];
  const int m0 = blockIdx.x * 16;
  stage_a_tile<2, N_DM>(h16 + (size_t)m0 * N_DM, shA);

  const int wave = threadIdx.x >> 5, lane = threadIdx.x & 31;
  const int n0 = blockIdx.y * 128 + wave * 16;
  const int which = blockIdx.z;
  const half_t* Bt = (which == 0) ? wqt : (which == 1) ? wkt : wvt;
  const float* bias = (which == 0) ? qb : (which == 1) ? kb : vb;
  v8f_t acc = wmma_rowtile_lds<N_DM>(shA, Bt, n0, lane);

  const int g = lane >> 4, nc = lane & 15;
  const int n = n0 + nc, h = n >> 6, kk = n & 63;
  const float bi = bias[n];
  if (which < 2) {
    half_t* out = (which == 0) ? q : k;
#pragma unroll
    for (int r = 0; r < 8; ++r) {
      const int t = m0 + r + 8 * g, bI = t >> 10, s = t & 1023;
      out[(((size_t)(bI * N_H + h)) * N_S + s) * N_DK + kk] = (half_t)(acc[r] + bi);
    }
  } else {
#pragma unroll
    for (int r = 0; r < 8; ++r) {
      const int t = m0 + r + 8 * g, bI = t >> 10, s = t & 1023;
      vt[(((size_t)(bI * N_H + h)) * N_DV + kk) * N_S + s] = (half_t)(acc[r] + bi);
    }
  }
}

// ---------------------------------------------------------------------------
// Scores: per (b,h)  q_bh[S,64] @ k_bh[S,64]^T * 1/8 + key mask -> wt (f32).
// grid (S/16, S/128, B*H), block 256, q tile staged in LDS.
// ---------------------------------------------------------------------------
__global__ __launch_bounds__(256) void k_scores(
    const half_t* __restrict__ q, const half_t* __restrict__ k,
    const unsigned char* __restrict__ mk, float* __restrict__ wt) {
  __shared__ alignas(32) char shA[PADDED_BYTES(16 * N_DK * 2)];
  const int bh = blockIdx.z, bI = bh >> 3;
  const half_t* qbh = q + (size_t)bh * N_S * N_DK;
  const half_t* kbh = k + (size_t)bh * N_S * N_DK;
  const int m0 = blockIdx.x * 16;
  stage_a_tile<2, N_DK>(qbh + (size_t)m0 * N_DK, shA);

  const int wave = threadIdx.x >> 5, lane = threadIdx.x & 31;
  const int n0 = blockIdx.y * 128 + wave * 16;
  v8f_t acc = wmma_rowtile_lds<N_DK>(shA, kbh, n0, lane);

  const int g = lane >> 4, nc = lane & 15, tcol = n0 + nc;
  const float maskv = mk[bI * N_S + tcol] ? 0.0f : -1e9f;
  float* out = wt + (size_t)bh * N_S * N_S;
#pragma unroll
  for (int r = 0; r < 8; ++r)
    out[(size_t)(m0 + r + 8 * g) * N_S + tcol] = acc[r] * 0.125f + maskv;
}

// ---------------------------------------------------------------------------
// Row softmax over S=1024, in place.  1 row / block, 256 threads.
// ---------------------------------------------------------------------------
__global__ __launch_bounds__(256) void k_softmax(float* __restrict__ wt) {
  __shared__ float sh[256];
  float* p = wt + (size_t)blockIdx.x * N_S;
  const int tid = threadIdx.x;
  v4f_t v = *(const v4f_t*)(p + tid * 4);
  float m = block_max(fmaxf(fmaxf(v[0], v[1]), fmaxf(v[2], v[3])), sh);
  v4f_t e;
  e[0] = __expf(v[0] - m); e[1] = __expf(v[1] - m);
  e[2] = __expf(v[2] - m); e[3] = __expf(v[3] - m);
  float s = block_sum(e[0] + e[1] + e[2] + e[3], sh);
  float inv = 1.0f / s;
  e[0] *= inv; e[1] *= inv; e[2] *= inv; e[3] *= inv;
  *(v4f_t*)(p + tid * 4) = e;
}

// ---------------------------------------------------------------------------
// Attention output: per (b,h)  probs[S,1024](f32, direct L2) @ vt[64,1024]^T
// -> head-concat cat [B,S,H*DV].  grid (S/16, 1, B*H), block 128 (4 waves).
// ---------------------------------------------------------------------------
__global__ __launch_bounds__(128) void k_attnout(
    const float* __restrict__ wt, const half_t* __restrict__ vt,
    half_t* __restrict__ cat) {
  const int bh = blockIdx.z, bI = bh >> 3, h = bh & 7;
  const float* A = wt + (size_t)bh * N_S * N_S;
  const half_t* Bt = vt + (size_t)bh * N_DV * N_S;
  const int m0 = blockIdx.x * 16;

  const int wave = threadIdx.x >> 5, lane = threadIdx.x & 31;
  const int n0 = wave * 16;
  v8f_t acc = wmma_rowtile_glb<true, N_S>(A + (size_t)m0 * N_S, Bt, n0, lane);

  const int g = lane >> 4, nc = lane & 15;
#pragma unroll
  for (int r = 0; r < 8; ++r) {
    const int s = m0 + r + 8 * g;
    cat[((size_t)(bI * N_S + s)) * (N_H * N_DV) + h * N_DV + n0 + nc] = (half_t)acc[r];
  }
}

// ---------------------------------------------------------------------------
// Generic GEMM -> f32 out with bias + residual.  N = 512, K templated.
// KK==512: LDS-staged A; KK==2048 (FFN2): direct A (keeps LDS small, high occ).
// grid (T/16, 4), block 256.
// ---------------------------------------------------------------------------
template <int KK>
__global__ __launch_bounds__(256) void k_gemm_bias_res_f32(
    const half_t* __restrict__ A, const half_t* __restrict__ Bt,
    const float* __restrict__ bias, const float* __restrict__ res,
    float* __restrict__ out) {
  const int m0 = blockIdx.x * 16;
  const int wave = threadIdx.x >> 5, lane = threadIdx.x & 31;
  const int n0 = blockIdx.y * 128 + wave * 16;
  v8f_t acc;
  if constexpr (KK == N_DM) {
    __shared__ alignas(32) char shA[PADDED_BYTES(16 * N_DM * 2)];
    stage_a_tile<2, N_DM>(A + (size_t)m0 * N_DM, shA);
    acc = wmma_rowtile_lds<N_DM>(shA, Bt, n0, lane);
  } else {
    acc = wmma_rowtile_glb<false, KK>(A + (size_t)m0 * KK, Bt, n0, lane);
  }

  const int g = lane >> 4, nc = lane & 15, n = n0 + nc;
  const float bi = bias[n];
#pragma unroll
  for (int r = 0; r < 8; ++r) {
    const size_t idx = (size_t)(m0 + r + 8 * g) * N_DM + n;
    out[idx] = res[idx] + acc[r] + bi;
  }
}

// ---------------------------------------------------------------------------
// FFN layer 1: [T,512] @ Bt[2048,512], ReLU, f16.  grid (T/16, 16), blk 256.
// ---------------------------------------------------------------------------
__global__ __launch_bounds__(256) void k_gemm_relu_f16(
    const half_t* __restrict__ A, const half_t* __restrict__ Bt,
    const float* __restrict__ bias, half_t* __restrict__ out) {
  __shared__ alignas(32) char shA[PADDED_BYTES(16 * N_DM * 2)];
  const int m0 = blockIdx.x * 16;
  stage_a_tile<2, N_DM>(A + (size_t)m0 * N_DM, shA);

  const int wave = threadIdx.x >> 5, lane = threadIdx.x & 31;
  const int n0 = blockIdx.y * 128 + wave * 16;
  v8f_t acc = wmma_rowtile_lds<N_DM>(shA, Bt, n0, lane);

  const int g = lane >> 4, nc = lane & 15, n = n0 + nc;
  const float bi = bias[n];
#pragma unroll
  for (int r = 0; r < 8; ++r)
    out[(size_t)(m0 + r + 8 * g) * N_DF + n] = (half_t)fmaxf(acc[r] + bi, 0.0f);
}

// ---------------------------------------------------------------------------
// Host-side orchestration
// ---------------------------------------------------------------------------
extern "C" void kernel_launch(void* const* d_in, const int* in_sizes, int n_in,
                              void* d_out, int out_size, void* d_ws, size_t ws_size,
                              hipStream_t stream) {
  (void)in_sizes; (void)n_in; (void)out_size; (void)ws_size;
  const float*         x  = (const float*)d_in[0];
  const unsigned char* mk = (const unsigned char*)d_in[1];
  const float* ln1_g = (const float*)d_in[2];
  const float* ln1_b = (const float*)d_in[3];
  const float* ln2_g = (const float*)d_in[4];
  const float* ln2_b = (const float*)d_in[5];
  const float* wq_w  = (const float*)d_in[6];
  const float* wq_b  = (const float*)d_in[7];
  const float* wk_w  = (const float*)d_in[8];
  const float* wk_b  = (const float*)d_in[9];
  const float* wv_w  = (const float*)d_in[10];
  const float* wv_b  = (const float*)d_in[11];
  const float* wo_w  = (const float*)d_in[12];
  const float* wo_b  = (const float*)d_in[13];
  const float* ffa_w = (const float*)d_in[14];
  const float* ffa_b = (const float*)d_in[15];
  const float* ffb_w = (const float*)d_in[16];
  const float* ffb_b = (const float*)d_in[17];

  float* out_x = (float*)d_out;                        // [T, DM]
  float* wt    = (float*)d_out + (size_t)N_T * N_DM;   // [B,H,S,S]

  char* ws = (char*)d_ws;
  size_t off = 0;
  auto take = [&](size_t bytes) -> char* {
    char* p = ws + off;
    off = (off + bytes + 255) & ~(size_t)255;
    return p;
  };
  half_t* wqt  = (half_t*)take((size_t)512 * 512 * 2);
  half_t* wkt  = (half_t*)take((size_t)512 * 512 * 2);
  half_t* wvt  = (half_t*)take((size_t)512 * 512 * 2);
  half_t* wot  = (half_t*)take((size_t)512 * 512 * 2);
  half_t* ffat = (half_t*)take((size_t)N_DF * 512 * 2);
  half_t* ffbt = (half_t*)take((size_t)512 * N_DF * 2);
  half_t* h16  = (half_t*)take((size_t)N_T * N_DM * 2);          // ln1 then ln2
  half_t* q16  = (half_t*)take((size_t)N_B * N_H * N_S * N_DK * 2);
  half_t* k16  = (half_t*)take((size_t)N_B * N_H * N_S * N_DK * 2);
  half_t* vt16 = (half_t*)take((size_t)N_B * N_H * N_DV * N_S * 2);
  half_t* cat16= (half_t*)take((size_t)N_T * N_H * N_DV * 2);
  float*  x1   = (float*)take((size_t)N_T * N_DM * 4);
  half_t* fmid = (half_t*)take((size_t)N_T * N_DF * 2);

  // 0) weight prep (f32 -> f16 B^T layouts)
  k_transpose_w<<<dim3(512), dim3(256), 0, stream>>>(wq_w, wqt, 512, 512, 1);
  k_transpose_w<<<dim3(512), dim3(256), 0, stream>>>(wk_w, wkt, 512, 512, 1);
  k_transpose_w<<<dim3(512), dim3(256), 0, stream>>>(wv_w, wvt, 512, 512, 1);
  k_transpose_w<<<dim3(512), dim3(256), 0, stream>>>(wo_w, wot, 512, 512, 0);
  k_transpose_w<<<dim3(2048), dim3(256), 0, stream>>>(ffa_w, ffat, 512, N_DF, 0);
  k_transpose_w<<<dim3(2048), dim3(256), 0, stream>>>(ffb_w, ffbt, N_DF, 512, 0);

  // 1) LN1
  k_layernorm16<<<dim3(N_T), dim3(128), 0, stream>>>(x, ln1_g, ln1_b, h16);

  // 2) QKV projections
  k_qkv<<<dim3(N_T / 16, 4, 3), dim3(256), 0, stream>>>(
      h16, wqt, wkt, wvt, wq_b, wk_b, wv_b, q16, k16, vt16);

  // 3) scores = q k^T / 8 + mask  (into wt region of d_out)
  k_scores<<<dim3(N_S / 16, N_S / 128, N_B * N_H), dim3(256), 0, stream>>>(
      q16, k16, mk, wt);

  // 4) softmax in place
  k_softmax<<<dim3(N_B * N_H * N_S), dim3(256), 0, stream>>>(wt);

  // 5) ot = wt @ v  -> head-concat
  k_attnout<<<dim3(N_S / 16, 1, N_B * N_H), dim3(128), 0, stream>>>(wt, vt16, cat16);

  // 6) x1 = x + cat @ wo + wo_b
  k_gemm_bias_res_f32<512><<<dim3(N_T / 16, 4), dim3(256), 0, stream>>>(
      cat16, wot, wo_b, x, x1);

  // 7) LN2
  k_layernorm16<<<dim3(N_T), dim3(128), 0, stream>>>(x1, ln2_g, ln2_b, h16);

  // 8) fmid = relu(h2 @ ffa + ffa_b)
  k_gemm_relu_f16<<<dim3(N_T / 16, 16), dim3(256), 0, stream>>>(
      h16, ffat, ffa_b, fmid);

  // 9) out = x1 + fmid @ ffb + ffb_b
  k_gemm_bias_res_f32<N_DF><<<dim3(N_T / 16, 4), dim3(256), 0, stream>>>(
      fmid, ffbt, ffb_b, x1, out_x);
}
